// DecoderWithoutCrossAttention_81028853007038
// MI455X (gfx1250) — matrix operand order repaired and checked by reading, not compiled, not run on hardware
//
#include <hip/hip_runtime.h>

#define BT 4096     // B*T rows
#define CE 384      // n_embd
#define HN 6        // heads
#define DH 64       // head dim
#define NL 6        // layers
#define FF 1536     // ffn hidden
#define VV 50257    // vocab
#define VP 50304    // vocab padded to 128
#define TT 256      // block size
#define NB 16       // batch

typedef __attribute__((ext_vector_type(4)))  float        floatx4;
typedef __attribute__((ext_vector_type(4)))  unsigned int uintx4;
typedef __attribute__((ext_vector_type(8)))  float        v8f;
typedef __attribute__((ext_vector_type(16))) __bf16       v16bf;

union Frag16 {
  unsigned short us[16];
  uintx4         u4[2];
  v16bf          v;
};

__device__ __forceinline__ unsigned short f2bf(float f) {
  union { float f; unsigned u; } x; x.f = f;
  unsigned r = x.u + 0x7fffu + ((x.u >> 16) & 1u);   // round-to-nearest-even
  return (unsigned short)(r >> 16);
}

__device__ __forceinline__ v8f zero8() {
  v8f z;
#pragma unroll
  for (int i = 0; i < 8; ++i) z[i] = 0.f;
  return z;
}

// Async copy of 32 contiguous bytes global->LDS per lane.
// ldsaddr = low 32 bits of the flat address of the LDS destination
// (== wave-relative LDS offset per ISA 10.2 aperture rules). The ptrtoint
// that produces it escapes the shared array, and the "memory" clobber orders
// the LDS reads after this asm.
__device__ __forceinline__ void async_copy32(unsigned ldsaddr, const unsigned short* ag) {
  asm volatile(
      "global_load_async_to_lds_b128 %0, %1, off\n\t"
      "global_load_async_to_lds_b128 %0, %1, off offset:16"
      :: "v"(ldsaddr), "v"(ag)
      : "memory");
}

// B-operand fragment swizzle: element (k,n) of a [K,N] bf16 matrix ->
// tile (n/16, k/32), lane = (k&31>=16)*16 + (n&15), elem = k&15 within lane.
__device__ __forceinline__ size_t bswz(int k, int n, int K) {
  int kk = k & 31;
  return ((size_t)(n >> 4) * (size_t)(K >> 5) + (size_t)(k >> 5)) * 512
       + (size_t)(((kk >> 4) << 4) + (n & 15)) * 16 + (kk & 15);
}

// ---------------------------------------------------------------------------
// Weight repack kernels (fp32 -> bf16, fragment-swizzled)
// ---------------------------------------------------------------------------
__global__ void repack_std_kernel(const float* __restrict__ src, unsigned short* __restrict__ dst,
                                  int K, int N, int layers) {
  long i = (long)blockIdx.x * 256 + threadIdx.x;
  long kn = (long)K * N;
  if (i >= layers * kn) return;
  long l = i / kn; long r = i % kn;
  int k = (int)(r / N); int n = (int)(r % N);
  dst[(size_t)l * kn + bswz(k, n, K)] = f2bf(src[i]);
}

// W [L][H][C][D] -> d [L][C*C swizzled] with n = h*D+dd, k = c
__global__ void repack_qkv_kernel(const float* __restrict__ W, unsigned short* __restrict__ d) {
  long i = (long)blockIdx.x * 256 + threadIdx.x;
  if (i >= (long)NL * CE * CE) return;
  int l  = (int)(i / (HN * CE * DH));
  int r  = (int)(i % (HN * CE * DH));
  int h  = r / (CE * DH);
  int r2 = r % (CE * DH);
  int c  = r2 / DH;
  int dd = r2 % DH;
  d[(size_t)l * CE * CE + bswz(c, h * DH + dd, CE)] = f2bf(W[i]);
}

// Wlm [C][V] -> swizzled [C][VP] (zero padded)
__global__ void repack_lm_kernel(const float* __restrict__ W, unsigned short* __restrict__ d) {
  long i = (long)blockIdx.x * 256 + threadIdx.x;
  if (i >= (long)CE * VP) return;
  int k = (int)(i / VP); int n = (int)(i % VP);
  d[bswz(k, n, CE)] = (n < VV) ? f2bf(W[(size_t)k * VV + n]) : (unsigned short)0;
}

// ---------------------------------------------------------------------------
// Embedding: x = tok[idx] + pos   (fp32, feeds LayerNorm)
// ---------------------------------------------------------------------------
__global__ void embed_kernel(const float* __restrict__ tok, const float* __restrict__ pos,
                             const int* __restrict__ idx, float* __restrict__ x) {
  int i = blockIdx.x * 256 + threadIdx.x;
  if (i >= BT * CE) return;
  int row = i / CE, c = i % CE;
  int t = row % TT;
  x[i] = tok[(size_t)idx[row] * CE + c] + pos[(size_t)t * CE + c];
}

// ---------------------------------------------------------------------------
// LayerNorm: one wave32 per row of 384; output bf16 row-major (GEMM A operand)
// ---------------------------------------------------------------------------
__global__ __launch_bounds__(256) void ln_kernel(const float* __restrict__ x,
                                                 const float* __restrict__ g,
                                                 const float* __restrict__ b,
                                                 unsigned short* __restrict__ y, int rows) {
  int wave = threadIdx.x >> 5, lane = threadIdx.x & 31;
  int row = blockIdx.x * 8 + wave;
  if (row >= rows) return;
  const float* xr = x + (size_t)row * CE;
  float vals[12]; float s = 0.f;
#pragma unroll
  for (int i = 0; i < 12; ++i) { vals[i] = xr[lane + i * 32]; s += vals[i]; }
#pragma unroll
  for (int m = 1; m < 32; m <<= 1) s += __shfl_xor(s, m, 32);
  float mu = s * (1.f / 384.f);
  float vr = 0.f;
#pragma unroll
  for (int i = 0; i < 12; ++i) { float d = vals[i] - mu; vr += d * d; }
#pragma unroll
  for (int m = 1; m < 32; m <<= 1) vr += __shfl_xor(vr, m, 32);
  float rstd = rsqrtf(vr * (1.f / 384.f) + 1e-5f);
  unsigned short* yr = y + (size_t)row * CE;
#pragma unroll
  for (int i = 0; i < 12; ++i) {
    int c = lane + i * 32;
    yr[c] = f2bf((vals[i] - mu) * rstd * g[c] + b[c]);
  }
}

// ---------------------------------------------------------------------------
// GEMM: acc = A[M,K](bf16 rm) * B[K,N](bf16 fragment-swizzled)
// epilogue: +bias, optional relu, optional +res(fp32); outputs any of:
//   Cf  fp32 row-major, Cb bf16 row-major, Cbt bf16 transposed [N][M]
// block = 256 threads (8 waves, 2x4), tile 128x128, K-step 32.
// A tile double-buffered in LDS via CDNA5 async global->LDS loads; B
// fragments double-buffered in registers; last K-step peeled so the main
// loop body is branch-free.
// ---------------------------------------------------------------------------
__global__ __launch_bounds__(256) void gemm_kernel(
    const unsigned short* __restrict__ A, const unsigned short* __restrict__ Bw,
    const float* __restrict__ bias, const float* __restrict__ res,
    float* __restrict__ Cf, unsigned short* __restrict__ Cb,
    unsigned short* __restrict__ Cbt,
    int M, int N, int K, int relu) {
  __shared__ __align__(16) unsigned short As[2][5120];  // [buf][row*40 + k]
  const int tid = threadIdx.x;
  const int m0 = blockIdx.y * 128, n0 = blockIdx.x * 128;
  const int w = tid >> 5, lane = tid & 31, lm = lane & 15, lh = lane >> 4;
  const int wm = w >> 2, wn = w & 3;
  const int Kt = K >> 5;

  v8f acc[4][2];
#pragma unroll
  for (int i = 0; i < 4; ++i)
#pragma unroll
    for (int j = 0; j < 2; ++j) acc[i][j] = zero8();

  // A staging: each thread covers 32 contiguous bytes of one row
  const int srow = tid >> 1, shalf = tid & 1;
  const unsigned short* Ag = A + (size_t)(m0 + srow) * K + shalf * 16;
  unsigned ldsb[2];
  ldsb[0] = (unsigned)(size_t)&As[0][srow * 40 + shalf * 16];
  ldsb[1] = (unsigned)(size_t)&As[1][srow * 40 + shalf * 16];

  // Visible stores to As: keeps the optimizer from treating the shared array
  // as store-free (the real writes happen through the async asm). The barrier
  // completes these before the first async write can land.
  As[0][srow * 40 + shalf * 16] = 0;
  As[1][srow * 40 + shalf * 16] = 0;
  __syncthreads();

  // per-wave B fragment base pointers (contiguous 1KB per wave per fragment)
  const unsigned short* nb[2];
#pragma unroll
  for (int j = 0; j < 2; ++j)
    nb[j] = Bw + ((size_t)((n0 >> 4) + wn * 2 + j) * (size_t)Kt) * 512 + lane * 16;

  // prologue: async-stage tile 0, load B fragments for tile 0
  async_copy32(ldsb[0], Ag);
  Frag16 bf[2];
#pragma unroll
  for (int j = 0; j < 2; ++j) {
    bf[j].u4[0] = *(const uintx4*)nb[j];
    bf[j].u4[1] = *(const uintx4*)(nb[j] + 8);
  }

  // main loop: kt = 0 .. Kt-2 (next-step loads are unconditional)
  for (int kt = 0; kt < Kt - 1; ++kt) {
    asm volatile("s_wait_asynccnt 0x0" ::: "memory");
    __syncthreads();  // As[kt&1] ready for everyone; prior reads of other buf done
    async_copy32(ldsb[(kt + 1) & 1], Ag + (kt + 1) * 32);
    if (kt + 2 < Kt) __builtin_prefetch(nb[0] + (size_t)(kt + 2) * 512, 0, 1);

    Frag16 af[4];
    const unsigned short* Ab = &As[kt & 1][0];
#pragma unroll
    for (int i = 0; i < 4; ++i) {
      int ab = (wm * 64 + i * 16 + lm) * 40 + lh * 16;
      af[i].u4[0] = *(const uintx4*)(Ab + ab);
      af[i].u4[1] = *(const uintx4*)(Ab + ab + 8);
    }
    Frag16 bn[2];
#pragma unroll
    for (int j = 0; j < 2; ++j) {
      const unsigned short* bp = nb[j] + (size_t)(kt + 1) * 512;
      bn[j].u4[0] = *(const uintx4*)bp;
      bn[j].u4[1] = *(const uintx4*)(bp + 8);
    }
#pragma unroll
    for (int j = 0; j < 2; ++j)
#pragma unroll
      for (int i = 0; i < 4; ++i)
        acc[i][j] = __builtin_amdgcn_wmma_f32_16x16x32_bf16(
            false, af[i].v, false, bf[j].v, (short)0, acc[i][j], false, false);
    bf[0] = bn[0]; bf[1] = bn[1];
  }

  // peeled last K-step
  {
    const int kt = Kt - 1;
    asm volatile("s_wait_asynccnt 0x0" ::: "memory");
    __syncthreads();
    Frag16 af[4];
    const unsigned short* Ab = &As[kt & 1][0];
#pragma unroll
    for (int i = 0; i < 4; ++i) {
      int ab = (wm * 64 + i * 16 + lm) * 40 + lh * 16;
      af[i].u4[0] = *(const uintx4*)(Ab + ab);
      af[i].u4[1] = *(const uintx4*)(Ab + ab + 8);
    }
#pragma unroll
    for (int j = 0; j < 2; ++j)
#pragma unroll
      for (int i = 0; i < 4; ++i)
        acc[i][j] = __builtin_amdgcn_wmma_f32_16x16x32_bf16(
            false, af[i].v, false, bf[j].v, (short)0, acc[i][j], false, false);
  }

#pragma unroll
  for (int j = 0; j < 2; ++j) {
    int col = n0 + (wn * 2 + j) * 16 + lm;
    if (col < N) {
      float bval = bias ? bias[col] : 0.f;
#pragma unroll
      for (int i = 0; i < 4; ++i)
#pragma unroll
        for (int r = 0; r < 8; ++r) {
          int row = m0 + wm * 64 + i * 16 + lh * 8 + r;
          float vv = acc[i][j][r] + bval;
          if (relu) vv = fmaxf(vv, 0.f);
          size_t o = (size_t)row * N + col;
          if (res) vv += res[o];
          if (Cf)  Cf[o] = vv;
          if (Cb)  Cb[o] = f2bf(vv);
          if (Cbt) Cbt[(size_t)col * M + row] = f2bf(vv);
        }
    }
  }
}

// ---------------------------------------------------------------------------
// Fused causal attention: one wave per (b, h, 16-row tile).
// q,k: bf16 row-major [BT][CE]; vt: bf16 transposed [CE][BT]; o: bf16 [BT][CE]
// ---------------------------------------------------------------------------
__global__ __launch_bounds__(32) void attn_kernel(
    const unsigned short* __restrict__ q, const unsigned short* __restrict__ k,
    const unsigned short* __restrict__ vt, unsigned short* __restrict__ o) {
  __shared__ __align__(16) unsigned short P[16 * 264];  // [row][s], stride 264
  const int mt = blockIdx.x;
  const int h  = blockIdx.y;
  const int b  = blockIdx.z;
  const int lane = threadIdx.x, lm = lane & 15, lh = lane >> 4;
  const int t0 = mt * 16;
  const int rowg = b * TT + t0;
  const int hc = h * DH;

  for (int i = lane; i < 16 * 264; i += 32) P[i] = 0;

  // Q fragments (K = 0..63, two 32-wide steps) straight from bf16 global
  Frag16 qa[2];
#pragma unroll
  for (int s = 0; s < 2; ++s) {
    const unsigned short* qp = q + (size_t)(rowg + lm) * CE + hc + s * 32 + lh * 16;
    qa[s].u4[0] = *(const uintx4*)qp;
    qa[s].u4[1] = *(const uintx4*)(qp + 8);
  }

  // scores for the non-masked column tiles (nt <= mt, uniform branch)
  v8f sacc[16];
#pragma unroll
  for (int nt = 0; nt < 16; ++nt) {
    if (nt <= mt) {
      v8f acc = zero8();
#pragma unroll
      for (int s = 0; s < 2; ++s) {
        Frag16 bfr;
        const unsigned short* kp =
            k + (size_t)(b * TT + nt * 16 + lm) * CE + hc + s * 32 + lh * 16;
        bfr.u4[0] = *(const uintx4*)kp;
        bfr.u4[1] = *(const uintx4*)(kp + 8);
        acc = __builtin_amdgcn_wmma_f32_16x16x32_bf16(
            false, qa[s].v, false, bfr.v, (short)0, acc, false, false);
      }
      sacc[nt] = acc;
    }
  }

  // causal mask + row softmax (rows: lanes 0-15 = r, lanes 16-31 = r+8)
  const float scale = 0.125f;  // 1/sqrt(64)
  float rmax[8], rsum[8];
#pragma unroll
  for (int r = 0; r < 8; ++r) rmax[r] = -3.4e38f;
#pragma unroll
  for (int nt = 0; nt < 16; ++nt) {
    if (nt <= mt) {
#pragma unroll
      for (int r = 0; r < 8; ++r) {
        int trow = t0 + lh * 8 + r;
        int scol = nt * 16 + lm;
        float val = sacc[nt][r] * scale;
        if (scol > trow) val = -__builtin_inff();
        sacc[nt][r] = val;
        rmax[r] = fmaxf(rmax[r], val);
      }
    }
  }
#pragma unroll
  for (int r = 0; r < 8; ++r) {
#pragma unroll
    for (int m = 1; m < 16; m <<= 1) rmax[r] = fmaxf(rmax[r], __shfl_xor(rmax[r], m, 32));
    rsum[r] = 0.f;
  }
#pragma unroll
  for (int nt = 0; nt < 16; ++nt) {
    if (nt <= mt) {
#pragma unroll
      for (int r = 0; r < 8; ++r) {
        float e = __expf(sacc[nt][r] - rmax[r]);
        sacc[nt][r] = e;
        rsum[r] += e;
      }
    }
  }
#pragma unroll
  for (int r = 0; r < 8; ++r) {
#pragma unroll
    for (int m = 1; m < 16; m <<= 1) rsum[r] += __shfl_xor(rsum[r], m, 32);
    rsum[r] = 1.f / rsum[r];
  }
#pragma unroll
  for (int nt = 0; nt < 16; ++nt) {
    if (nt <= mt) {
#pragma unroll
      for (int r = 0; r < 8; ++r)
        P[(lh * 8 + r) * 264 + nt * 16 + lm] = f2bf(sacc[nt][r] * rsum[r]);
    }
  }
  __syncthreads();

  // O = P @ V  (K only over the unmasked range, rounded up to 32)
  const int Kend = ((mt + 1) * 16 + 31) & ~31;
  v8f oacc[4];
#pragma unroll
  for (int i = 0; i < 4; ++i) oacc[i] = zero8();
#pragma unroll
  for (int ks = 0; ks < 8; ++ks) {
    int kk0 = ks * 32;
    if (kk0 < Kend) {
      Frag16 pa;
      int ab = lm * 264 + kk0 + lh * 16;
      pa.u4[0] = *(const uintx4*)(P + ab);
      pa.u4[1] = *(const uintx4*)(P + ab + 8);
#pragma unroll
      for (int nt = 0; nt < 4; ++nt) {
        Frag16 vbf;
        const unsigned short* vp =
            vt + (size_t)(hc + nt * 16 + lm) * BT + b * TT + kk0 + lh * 16;
        vbf.u4[0] = *(const uintx4*)vp;
        vbf.u4[1] = *(const uintx4*)(vp + 8);
        oacc[nt] = __builtin_amdgcn_wmma_f32_16x16x32_bf16(
            false, pa.v, false, vbf.v, (short)0, oacc[nt], false, false);
      }
    }
  }
#pragma unroll
  for (int nt = 0; nt < 4; ++nt)
#pragma unroll
    for (int r = 0; r < 8; ++r)
      o[(size_t)(rowg + lh * 8 + r) * CE + hc + nt * 16 + lm] = f2bf(oacc[nt][r]);
}

// ---------------------------------------------------------------------------
// Loss: per-row logsumexp NLL, then mean
// ---------------------------------------------------------------------------
__global__ __launch_bounds__(256) void nll_kernel(const float* __restrict__ logits,
                                                  const int* __restrict__ tgt,
                                                  float* __restrict__ nll) {
  __shared__ float red[256];
  int row = blockIdx.x, tid = threadIdx.x;
  const float* lr = logits + (size_t)row * VV;
  float m = -3.4e38f;
  for (int i = tid; i < VV; i += 256) m = fmaxf(m, lr[i]);
  red[tid] = m; __syncthreads();
  for (int s = 128; s > 0; s >>= 1) { if (tid < s) red[tid] = fmaxf(red[tid], red[tid + s]); __syncthreads(); }
  m = red[0]; __syncthreads();
  float sm = 0.f;
  for (int i = tid; i < VV; i += 256) sm += __expf(lr[i] - m);
  red[tid] = sm; __syncthreads();
  for (int s = 128; s > 0; s >>= 1) { if (tid < s) red[tid] += red[tid + s]; __syncthreads(); }
  if (tid == 0) nll[row] = (logf(red[0]) + m) - lr[tgt[row]];
}

__global__ __launch_bounds__(256) void mean_kernel(const float* __restrict__ nll,
                                                   float* __restrict__ out) {
  __shared__ float red[256];
  float s = 0.f;
  for (int i = threadIdx.x; i < BT; i += 256) s += nll[i];
  red[threadIdx.x] = s; __syncthreads();
  for (int t = 128; t > 0; t >>= 1) { if (threadIdx.x < t) red[threadIdx.x] += red[threadIdx.x + t]; __syncthreads(); }
  if (threadIdx.x == 0) out[0] = red[0] * (1.f / BT);
}

// ---------------------------------------------------------------------------
extern "C" void kernel_launch(void* const* d_in, const int* in_sizes, int n_in,
                              void* d_out, int out_size, void* d_ws, size_t ws_size,
                              hipStream_t stream) {
  const float* tok  = (const float*)d_in[0];
  const float* pos  = (const float*)d_in[1];
  const float* Wq   = (const float*)d_in[2];
  const float* Wk   = (const float*)d_in[3];
  const float* Wv   = (const float*)d_in[4];
  const float* Wp   = (const float*)d_in[5];
  const float* bp   = (const float*)d_in[6];
  const float* W1   = (const float*)d_in[7];
  const float* b1   = (const float*)d_in[8];
  const float* W2   = (const float*)d_in[9];
  const float* b2   = (const float*)d_in[10];
  const float* ln1g = (const float*)d_in[11];
  const float* ln1b = (const float*)d_in[12];
  const float* ln2g = (const float*)d_in[13];
  const float* ln2b = (const float*)d_in[14];
  const float* lng  = (const float*)d_in[15];
  const float* lnb  = (const float*)d_in[16];
  const float* Wlm  = (const float*)d_in[17];
  const float* blm  = (const float*)d_in[18];
  const int* idx    = (const int*)d_in[19];
  const int* target = (const int*)d_in[20];
  float* out = (float*)d_out;

  char* base = (char*)d_ws; size_t off = 0;
  auto alloc = [&](size_t bytes) -> void* {
    void* p = base + off; off = (off + bytes + 255) & ~(size_t)255; return p;
  };
  float*          x     = (float*)alloc((size_t)BT * CE * 4);
  unsigned short* hbf   = (unsigned short*)alloc((size_t)BT * CE * 2);
  unsigned short* qbf   = (unsigned short*)alloc((size_t)BT * CE * 2);
  unsigned short* kbf   = (unsigned short*)alloc((size_t)BT * CE * 2);
  unsigned short* vtbf  = (unsigned short*)alloc((size_t)CE * BT * 2);
  unsigned short* obf   = (unsigned short*)alloc((size_t)BT * CE * 2);
  unsigned short* ffhbf = (unsigned short*)alloc((size_t)BT * FF * 2);
  float*          nllb  = (float*)alloc((size_t)BT * 4);
  unsigned short* wqkv  = (unsigned short*)alloc((size_t)3 * NL * CE * CE * 2);
  unsigned short* wpb   = (unsigned short*)alloc((size_t)NL * CE * CE * 2);
  unsigned short* w1b   = (unsigned short*)alloc((size_t)NL * CE * FF * 2);
  unsigned short* w2b   = (unsigned short*)alloc((size_t)NL * FF * CE * 2);
  unsigned short* wlmb  = (unsigned short*)alloc((size_t)CE * VP * 2);

  auto cdiv = [](long a, long b) { return (int)((a + b - 1) / b); };

  repack_qkv_kernel<<<cdiv((long)NL * CE * CE, 256), 256, 0, stream>>>(Wq, wqkv);
  repack_qkv_kernel<<<cdiv((long)NL * CE * CE, 256), 256, 0, stream>>>(Wk, wqkv + (size_t)NL * CE * CE);
  repack_qkv_kernel<<<cdiv((long)NL * CE * CE, 256), 256, 0, stream>>>(Wv, wqkv + (size_t)2 * NL * CE * CE);
  repack_std_kernel<<<cdiv((long)NL * CE * CE, 256), 256, 0, stream>>>(Wp, wpb, CE, CE, NL);
  repack_std_kernel<<<cdiv((long)NL * CE * FF, 256), 256, 0, stream>>>(W1, w1b, CE, FF, NL);
  repack_std_kernel<<<cdiv((long)NL * FF * CE, 256), 256, 0, stream>>>(W2, w2b, FF, CE, NL);
  repack_lm_kernel<<<cdiv((long)CE * VP, 256), 256, 0, stream>>>(Wlm, wlmb);

  embed_kernel<<<cdiv((long)BT * CE, 256), 256, 0, stream>>>(tok, pos, idx, x);

  auto gemm = [&](const unsigned short* A, const unsigned short* Bw, const float* bias,
                  const float* res, float* Cf, unsigned short* Cb, unsigned short* Cbt,
                  int M_, int N_, int K_, int relu) {
    dim3 g((N_ + 127) / 128, M_ / 128);
    gemm_kernel<<<g, 256, 0, stream>>>(A, Bw, bias, res, Cf, Cb, Cbt, M_, N_, K_, relu);
  };

  for (int l = 0; l < NL; ++l) {
    ln_kernel<<<BT / 8, 256, 0, stream>>>(x, ln1g + l * CE, ln1b + l * CE, hbf, BT);
    gemm(hbf, wqkv + (size_t)l * CE * CE,               nullptr, nullptr, nullptr, qbf, nullptr, BT, CE, CE, 0);
    gemm(hbf, wqkv + (size_t)(NL + l) * CE * CE,        nullptr, nullptr, nullptr, kbf, nullptr, BT, CE, CE, 0);
    gemm(hbf, wqkv + (size_t)(2 * NL + l) * CE * CE,    nullptr, nullptr, nullptr, nullptr, vtbf, BT, CE, CE, 0);
    attn_kernel<<<dim3(TT / 16, HN, NB), 32, 0, stream>>>(qbf, kbf, vtbf, obf);
    gemm(obf, wpb + (size_t)l * CE * CE, bp + (size_t)l * CE, x, x, nullptr, nullptr, BT, CE, CE, 0);
    ln_kernel<<<BT / 8, 256, 0, stream>>>(x, ln2g + l * CE, ln2b + l * CE, hbf, BT);
    gemm(hbf, w1b + (size_t)l * CE * FF, b1 + (size_t)l * FF, nullptr, nullptr, ffhbf, nullptr, BT, FF, CE, 1);
    gemm(ffhbf, w2b + (size_t)l * FF * CE, b2 + (size_t)l * CE, x, x, nullptr, nullptr, BT, CE, FF, 0);
  }
  ln_kernel<<<BT / 8, 256, 0, stream>>>(x, lng, lnb, hbf, BT);
  gemm(hbf, wlmb, blm, nullptr, out, nullptr, nullptr, BT, VV, CE, 0);

  nll_kernel<<<BT, 256, 0, stream>>>(out, target, nllb);
  mean_kernel<<<1, 256, 0, stream>>>(nllb, out + (size_t)out_size - 1);

  (void)in_sizes; (void)n_in; (void)ws_size;
}
